// TimeSeriesCnnEmbedding_80178449482410
// MI455X (gfx1250) — matrix operand order repaired and checked by reading, not compiled, not run on hardware
//
#include <hip/hip_runtime.h>
#include <hip/hip_bf16.h>

typedef __attribute__((ext_vector_type(16))) _Float16 v16h;
typedef __attribute__((ext_vector_type(8)))  _Float16 v8h;
typedef __attribute__((ext_vector_type(8)))  float    v8f;

#define KSZ   64
#define BETA  0.25f
#define VOCAB 512
#define EDIM  256

// ---------------------------------------------------------------------------
// Implicit-GEMM conv: out[b,co,t] = sum_{ci,k} w[co,ci,k]*in[b,ci,2t+k] + bias
// GEMM: M=Cout, N=B*Tout, K=Cin*64.  f16 WMMA / f32 accum.
// Block = 256 threads (8 waves). Block tile 64M x 128N; wave tile 32M x 32N
// (2x2 WMMA). LDS tiles stored in WMMA *fragment order* (32B contiguous per
// lane -> ds_read_b128 fragment loads), double-buffered.
// ---------------------------------------------------------------------------
__global__ __launch_bounds__(256)
void conv_wmma_kernel(const float* __restrict__ in, const float* __restrict__ w,
                      const float* __restrict__ bias, float* __restrict__ out,
                      int Cin, int Cout, int Tin, int Tout, int Ntot)
{
    // fragment-order tiles: [buf][frag][lane][elem]
    __shared__ alignas(32) _Float16 As[2][4][32][16];   //  8 KB
    __shared__ alignas(32) _Float16 Bs[2][8][32][16];   // 16 KB

    const int tid = threadIdx.x;
    const int co0 = blockIdx.x * 64;
    const int n0  = blockIdx.y * 128;

    // ---- A staging assignment: 2048 halfs / 256 thr = 8 each ----
    const int afrag = tid >> 6;               // 0..3
    const int alane = (tid >> 1) & 31;
    const int ae0   = (tid & 1) * 8;          // element 0 or 8
    const int am    = co0 + afrag * 16 + (alane & 15);
    const int akb   = (ae0 ? 8 : 0) + 16 * (alane >> 4);  // contiguous k base

    // ---- B staging assignment: 4096 halfs / 256 thr = 16 each ----
    const int bfrag = tid >> 5;               // 0..7
    const int blane = tid & 31;
    const int bh    = blane >> 4;
    const int bn    = n0 + bfrag * 16 + (blane & 15);
    const bool bvalid = (bn < Ntot);
    int bb = 0, bt = 0;
    if (bvalid) { bb = bn / Tout; bt = bn - bb * Tout; }
    const size_t inBase = (size_t)bb * Cin * Tin + 2 * bt;

    // ---- compute assignment: 8 waves = 2(M) x 4(N) ----
    const int lane = tid & 31, wv = tid >> 5;
    const int mq = wv >> 2, nq = wv & 3;
    const int half_ = lane >> 4;

    const v8f zero8 = {0.f,0.f,0.f,0.f,0.f,0.f,0.f,0.f};
    v8f acc[2][2] = {{zero8, zero8}, {zero8, zero8}};

    const int nChunks = Cin * 2;              // K = Cin*64, chunk = 32

    auto stageA = [&](int buf, int c) {
        const int ci = c >> 1, kk0 = (c & 1) * 32;
        const float* wp = w + (size_t)am * (Cin * KSZ) + ci * KSZ + kk0 + akb;
        v8h t;
        #pragma unroll
        for (int i = 0; i < 8; ++i) t[i] = (_Float16)wp[i];
        *(v8h*)&As[buf][afrag][alane][ae0] = t;
    };
    auto stageB = [&](int buf, int c) {
        const int ci = c >> 1, kk0 = (c & 1) * 32;
        const float* ip = in + inBase + (size_t)ci * Tin + kk0 + 16 * bh;
        v16h t;
        #pragma unroll
        for (int i = 0; i < 16; ++i) t[i] = (_Float16)(bvalid ? ip[i] : 0.f);
        *(v16h*)&Bs[buf][bfrag][blane][0] = t;
        if (bvalid && c + 2 < nChunks) {
            const int c2 = c + 2, ci2 = c2 >> 1, kk2 = (c2 & 1) * 32;
            __builtin_prefetch(in + inBase + (size_t)ci2 * Tin + kk2 + 16 * bh, 0, 0);
        }
    };

    stageA(0, 0);
    stageB(0, 0);
    for (int c = 0; c < nChunks; ++c) {
        const int buf = c & 1;
        __syncthreads();
        if (c + 1 < nChunks) { stageA(buf ^ 1, c + 1); stageB(buf ^ 1, c + 1); }
        const v16h a0 = *(const v16h*)&As[buf][mq * 2 + 0][lane][0];
        const v16h a1 = *(const v16h*)&As[buf][mq * 2 + 1][lane][0];
        const v16h b0 = *(const v16h*)&Bs[buf][nq * 2 + 0][lane][0];
        const v16h b1 = *(const v16h*)&Bs[buf][nq * 2 + 1][lane][0];
        acc[0][0] = __builtin_amdgcn_wmma_f32_16x16x32_f16(false, a0, false, b0, (short)0, acc[0][0], false, false);
        acc[0][1] = __builtin_amdgcn_wmma_f32_16x16x32_f16(false, a0, false, b1, (short)0, acc[0][1], false, false);
        acc[1][0] = __builtin_amdgcn_wmma_f32_16x16x32_f16(false, a1, false, b0, (short)0, acc[1][0], false, false);
        acc[1][1] = __builtin_amdgcn_wmma_f32_16x16x32_f16(false, a1, false, b1, (short)0, acc[1][1], false, false);
    }

    #pragma unroll
    for (int ni = 0; ni < 2; ++ni) {
        const int nst = n0 + (nq * 2 + ni) * 16 + (lane & 15);
        if (nst < Ntot) {
            const int b2 = nst / Tout, t2 = nst - b2 * Tout;
            #pragma unroll
            for (int mi = 0; mi < 2; ++mi) {
                #pragma unroll
                for (int r = 0; r < 8; ++r) {
                    const int co = co0 + (mq * 2 + mi) * 16 + r + 8 * half_;
                    out[(size_t)b2 * Cout * Tout + (size_t)co * Tout + t2] =
                        acc[mi][ni][r] + bias[co];
                }
            }
        }
    }
}

// ---------------------------------------------------------------------------
// VQ distance GEMM: S[n,v] = ||c_v||^2 - 2 <z_n, c_v>.
// M = VOCAB (64-tiles), N = B*T4, K = EDIM (8 chunks of 32). Same scheme.
// ---------------------------------------------------------------------------
__global__ __launch_bounds__(256)
void vq_gemm_kernel(const float* __restrict__ cb, const float* __restrict__ h,
                    const float* __restrict__ cn, float* __restrict__ S,
                    int T, int Ntot)
{
    __shared__ alignas(32) _Float16 As[2][4][32][16];
    __shared__ alignas(32) _Float16 Bs[2][8][32][16];

    const int tid = threadIdx.x;
    const int v0  = blockIdx.x * 64;
    const int n0  = blockIdx.y * 128;

    const int afrag = tid >> 6;
    const int alane = (tid >> 1) & 31;
    const int ae0   = (tid & 1) * 8;
    const int am    = v0 + afrag * 16 + (alane & 15);
    const int akb   = (ae0 ? 8 : 0) + 16 * (alane >> 4);

    const int bfrag = tid >> 5;
    const int blane = tid & 31;
    const int bh    = blane >> 4;
    const int bn    = n0 + bfrag * 16 + (blane & 15);
    const bool bvalid = (bn < Ntot);
    int bb = 0, bt = 0;
    if (bvalid) { bb = bn / T; bt = bn - bb * T; }
    const size_t hBase = (size_t)bb * EDIM * T + bt;

    const int lane = tid & 31, wv = tid >> 5;
    const int mq = wv >> 2, nq = wv & 3;
    const int half_ = lane >> 4;

    const v8f zero8 = {0.f,0.f,0.f,0.f,0.f,0.f,0.f,0.f};
    v8f acc[2][2] = {{zero8, zero8}, {zero8, zero8}};

    auto stageA = [&](int buf, int c) {
        const int k0 = c * 32;
        const float* cp = cb + (size_t)am * EDIM + k0 + akb;
        v8h t;
        #pragma unroll
        for (int i = 0; i < 8; ++i) t[i] = (_Float16)cp[i];
        *(v8h*)&As[buf][afrag][alane][ae0] = t;
    };
    auto stageB = [&](int buf, int c) {
        const int k0 = c * 32 + 16 * bh;
        v16h t;
        #pragma unroll
        for (int i = 0; i < 16; ++i)
            t[i] = (_Float16)(bvalid ? h[hBase + (size_t)(k0 + i) * T] : 0.f);
        *(v16h*)&Bs[buf][bfrag][blane][0] = t;
    };

    stageA(0, 0);
    stageB(0, 0);
    for (int c = 0; c < EDIM / 32; ++c) {     // 8 chunks
        const int buf = c & 1;
        __syncthreads();
        if (c + 1 < EDIM / 32) { stageA(buf ^ 1, c + 1); stageB(buf ^ 1, c + 1); }
        const v16h a0 = *(const v16h*)&As[buf][mq * 2 + 0][lane][0];
        const v16h a1 = *(const v16h*)&As[buf][mq * 2 + 1][lane][0];
        const v16h b0 = *(const v16h*)&Bs[buf][nq * 2 + 0][lane][0];
        const v16h b1 = *(const v16h*)&Bs[buf][nq * 2 + 1][lane][0];
        acc[0][0] = __builtin_amdgcn_wmma_f32_16x16x32_f16(false, a0, false, b0, (short)0, acc[0][0], false, false);
        acc[0][1] = __builtin_amdgcn_wmma_f32_16x16x32_f16(false, a0, false, b1, (short)0, acc[0][1], false, false);
        acc[1][0] = __builtin_amdgcn_wmma_f32_16x16x32_f16(false, a1, false, b0, (short)0, acc[1][0], false, false);
        acc[1][1] = __builtin_amdgcn_wmma_f32_16x16x32_f16(false, a1, false, b1, (short)0, acc[1][1], false, false);
    }

    #pragma unroll
    for (int ni = 0; ni < 2; ++ni) {
        const int nst = n0 + (nq * 2 + ni) * 16 + (lane & 15);
        if (nst < Ntot) {
            #pragma unroll
            for (int mi = 0; mi < 2; ++mi) {
                #pragma unroll
                for (int r = 0; r < 8; ++r) {
                    const int vq = v0 + (mq * 2 + mi) * 16 + r + 8 * half_;
                    S[(size_t)nst * VOCAB + vq] = cn[vq] - 2.f * acc[mi][ni][r];
                }
            }
        }
    }
}

// ---------------------------------------------------------------------------
// BatchNorm statistics: one block per channel; biased variance over (B, T).
// ---------------------------------------------------------------------------
__global__ __launch_bounds__(256)
void bn_stats_kernel(const float* __restrict__ x, float* __restrict__ stats,
                     int C, int T, int Bsz)
{
    __shared__ float ss[256], ss2[256];
    int c = blockIdx.x, tid = threadIdx.x;
    float s = 0.f, s2 = 0.f;
    for (int b = 0; b < Bsz; ++b) {
        const float* p = x + (size_t)b * C * T + (size_t)c * T;
        for (int t = tid; t < T; t += 256) {
            float v = p[t];
            s += v; s2 += v * v;
        }
    }
    ss[tid] = s; ss2[tid] = s2;
    __syncthreads();
    for (int o = 128; o > 0; o >>= 1) {
        if (tid < o) { ss[tid] += ss[tid + o]; ss2[tid] += ss2[tid + o]; }
        __syncthreads();
    }
    if (tid == 0) {
        float BT  = (float)(Bsz * T);
        float mu  = ss[0] / BT;
        float var = ss2[0] / BT - mu * mu;
        stats[c]     = mu;
        stats[C + c] = rsqrtf(var + 1e-5f);
    }
}

// ---------------------------------------------------------------------------
// BN apply + PReLU, in place. grid = (ceil(T/256), B*C).
// ---------------------------------------------------------------------------
__global__ __launch_bounds__(256)
void bn_prelu_kernel(float* __restrict__ x, const float* __restrict__ stats,
                     const float* __restrict__ g, const float* __restrict__ be,
                     const float* __restrict__ a, int C, int T)
{
    int bc = blockIdx.y;
    int c  = bc % C;
    int t  = blockIdx.x * 256 + threadIdx.x;
    if (t >= T) return;
    size_t i = (size_t)bc * T + t;
    float al = a[0];
    float y = (x[i] - stats[c]) * stats[C + c] * g[c] + be[c];
    x[i] = (y >= 0.f) ? y : al * y;
}

// ---------------------------------------------------------------------------
__global__ __launch_bounds__(256)
void cnorm_kernel(const float* __restrict__ cb, float* __restrict__ cn)
{
    __shared__ float ss[256];
    int vv = blockIdx.x, tid = threadIdx.x;
    float v = cb[(size_t)vv * EDIM + tid];
    ss[tid] = v * v;
    __syncthreads();
    for (int o = 128; o > 0; o >>= 1) { if (tid < o) ss[tid] += ss[tid + o]; __syncthreads(); }
    if (tid == 0) cn[vv] = ss[0];
}

__global__ __launch_bounds__(256)
void znorm_kernel(const float* __restrict__ h, float* __restrict__ zn, int T)
{
    __shared__ float ss[256];
    int nidx = blockIdx.x, tid = threadIdx.x;
    int b = nidx / T, t = nidx - b * T;
    float v = h[(size_t)b * EDIM * T + (size_t)tid * T + t];
    ss[tid] = v * v;
    __syncthreads();
    for (int o = 128; o > 0; o >>= 1) { if (tid < o) ss[tid] += ss[tid + o]; __syncthreads(); }
    if (tid == 0) zn[nidx] = ss[0];
}

__global__ void init_counts_kernel(int* __restrict__ counts)
{
    counts[threadIdx.x] = 0;   // <<<1, VOCAB>>>
}

// ---------------------------------------------------------------------------
// Per-vector argmin (first-min tie-break like jnp.argmin), dmin, idx, z_q
// gather (transposed), one-hot encodings.
// ---------------------------------------------------------------------------
__global__ __launch_bounds__(256)
void argmin_kernel(const float* __restrict__ S, const float* __restrict__ zn,
                   const float* __restrict__ cb, int* __restrict__ counts,
                   float* __restrict__ dmin, float* __restrict__ zq,
                   float* __restrict__ enc, float* __restrict__ idxf, int T)
{
    __shared__ float bv[256];
    __shared__ int   bi[256];
    __shared__ int   sVmin;
    int nidx = blockIdx.x, tid = threadIdx.x;
    const float* row = S + (size_t)nidx * VOCAB;
    float v1 = row[tid], v2 = row[tid + 256];
    float best = v1; int bidx = tid;
    if (v2 < best) { best = v2; bidx = tid + 256; }
    bv[tid] = best; bi[tid] = bidx;
    __syncthreads();
    for (int o = 128; o > 0; o >>= 1) {
        if (tid < o) {
            if (bv[tid + o] < bv[tid] ||
                (bv[tid + o] == bv[tid] && bi[tid + o] < bi[tid])) {
                bv[tid] = bv[tid + o]; bi[tid] = bi[tid + o];
            }
        }
        __syncthreads();
    }
    if (tid == 0) {
        int vm = bi[0];
        sVmin = vm;
        dmin[nidx] = zn[nidx] + bv[0];
        idxf[nidx] = (float)vm;
        atomicAdd(&counts[vm], 1);
    }
    __syncthreads();
    int vm = sVmin;
    int b = nidx / T, t = nidx - b * T;
    zq[(size_t)b * EDIM * T + (size_t)tid * T + t] = cb[(size_t)vm * EDIM + tid];
    enc[(size_t)nidx * VOCAB + tid]       = (tid == vm)       ? 1.f : 0.f;
    enc[(size_t)nidx * VOCAB + tid + 256] = (tid + 256 == vm) ? 1.f : 0.f;
}

// ---------------------------------------------------------------------------
// Deterministic finalize: loss = (1+beta)*sum(dmin)/(N*EDIM);
// perplexity = exp(-sum(e_mean*log(e_mean+1e-10))).
// ---------------------------------------------------------------------------
__global__ __launch_bounds__(256)
void finalize_kernel(const float* __restrict__ dmin, const int* __restrict__ counts,
                     float* __restrict__ lossOut, float* __restrict__ perpOut, int N)
{
    __shared__ float ss[256];
    int tid = threadIdx.x;
    float s = 0.f;
    for (int i = tid; i < N; i += 256) s += dmin[i];
    ss[tid] = s;
    __syncthreads();
    for (int o = 128; o > 0; o >>= 1) { if (tid < o) ss[tid] += ss[tid + o]; __syncthreads(); }
    float loss = 0.f;
    if (tid == 0) loss = (1.f + BETA) * ss[0] / ((float)N * (float)EDIM);
    __syncthreads();
    float hsum = 0.f;
    for (int v = tid; v < VOCAB; v += 256) {
        float em = (float)counts[v] / (float)N;
        hsum += em * logf(em + 1e-10f);
    }
    ss[tid] = hsum;
    __syncthreads();
    for (int o = 128; o > 0; o >>= 1) { if (tid < o) ss[tid] += ss[tid + o]; __syncthreads(); }
    if (tid == 0) { lossOut[0] = loss; perpOut[0] = expf(-ss[0]); }
}

// ---------------------------------------------------------------------------
extern "C" void kernel_launch(void* const* d_in, const int* in_sizes, int n_in,
                              void* d_out, int out_size, void* d_ws, size_t ws_size,
                              hipStream_t stream)
{
    (void)in_sizes; (void)n_in; (void)out_size; (void)ws_size;
    const float* x   = (const float*)d_in[0];
    const float* w1  = (const float*)d_in[1];
    const float* b1  = (const float*)d_in[2];
    const float* g1  = (const float*)d_in[3];
    const float* be1 = (const float*)d_in[4];
    const float* a1  = (const float*)d_in[5];
    const float* w2  = (const float*)d_in[6];
    const float* b2  = (const float*)d_in[7];
    const float* g2  = (const float*)d_in[8];
    const float* be2 = (const float*)d_in[9];
    const float* a2  = (const float*)d_in[10];
    const float* w3  = (const float*)d_in[11];
    const float* b3  = (const float*)d_in[12];
    const float* g3  = (const float*)d_in[13];
    const float* be3 = (const float*)d_in[14];
    const float* a3  = (const float*)d_in[15];
    const float* w4  = (const float*)d_in[16];
    const float* b4  = (const float*)d_in[17];
    const float* g4  = (const float*)d_in[18];
    const float* be4 = (const float*)d_in[19];
    const float* a4  = (const float*)d_in[20];
    const float* cb  = (const float*)d_in[21];

    const int Bsz = 32;
    const int T0 = 16384, T1 = 8161, T2 = 4049, T3 = 1993, T4 = 965;
    const int C1 = 64, C2 = 128, C3 = 256, C4 = 256;
    const int N1 = Bsz * T1, N2 = Bsz * T2, N3 = Bsz * T3, N4 = Bsz * T4;  // N4 = 30880

    // ---- workspace layout (floats), aliased buffers ----
    float* ws = (float*)d_ws;
    const size_t szA = (size_t)Bsz * C1 * T1;   // 16,713,728 >= act3, >= S
    const size_t szB = (size_t)Bsz * C2 * T2;   // 16,584,704 >= act4
    float* act1  = ws;                 // bufA
    float* act2  = ws + szA;           // bufB
    float* act3  = ws;                 // bufA (act1 dead after conv2)
    float* act4  = ws + szA;           // bufB (act2 dead after conv3)
    float* Smat  = ws;                 // bufA (act3 dead after conv4)
    float* stats = ws + szA + szB;     // 512
    float* cn    = stats + 512;        // 512
    float* zn    = cn + 512;           // N4
    float* dmin  = zn + N4;            // N4
    int*   counts = (int*)(dmin + N4); // 512 ints

    // ---- output layout: loss | z_q | perplexity | min_encodings | idx ----
    float* outLoss = (float*)d_out;
    float* outZq   = outLoss + 1;
    float* outPerp = outZq + (size_t)Bsz * C4 * T4;
    float* outEnc  = outPerp + 1;
    float* outIdx  = outEnc + (size_t)N4 * VOCAB;

    const dim3 blk(256);

    // layer 1
    conv_wmma_kernel<<<dim3(C1 / 64, (N1 + 127) / 128), blk, 0, stream>>>(
        x, w1, b1, act1, 1, C1, T0, T1, N1);
    bn_stats_kernel<<<C1, 256, 0, stream>>>(act1, stats, C1, T1, Bsz);
    bn_prelu_kernel<<<dim3((T1 + 255) / 256, Bsz * C1), 256, 0, stream>>>(
        act1, stats, g1, be1, a1, C1, T1);
    // layer 2
    conv_wmma_kernel<<<dim3(C2 / 64, (N2 + 127) / 128), blk, 0, stream>>>(
        act1, w2, b2, act2, C1, C2, T1, T2, N2);
    bn_stats_kernel<<<C2, 256, 0, stream>>>(act2, stats, C2, T2, Bsz);
    bn_prelu_kernel<<<dim3((T2 + 255) / 256, Bsz * C2), 256, 0, stream>>>(
        act2, stats, g2, be2, a2, C2, T2);
    // layer 3
    conv_wmma_kernel<<<dim3(C3 / 64, (N3 + 127) / 128), blk, 0, stream>>>(
        act2, w3, b3, act3, C2, C3, T2, T3, N3);
    bn_stats_kernel<<<C3, 256, 0, stream>>>(act3, stats, C3, T3, Bsz);
    bn_prelu_kernel<<<dim3((T3 + 255) / 256, Bsz * C3), 256, 0, stream>>>(
        act3, stats, g3, be3, a3, C3, T3);
    // layer 4
    conv_wmma_kernel<<<dim3(C4 / 64, (N4 + 127) / 128), blk, 0, stream>>>(
        act3, w4, b4, act4, C3, C4, T3, T4, N4);
    bn_stats_kernel<<<C4, 256, 0, stream>>>(act4, stats, C4, T4, Bsz);
    bn_prelu_kernel<<<dim3((T4 + 255) / 256, Bsz * C4), 256, 0, stream>>>(
        act4, stats, g4, be4, a4, C4, T4);

    // ---- vector quantizer ----
    cnorm_kernel<<<VOCAB, 256, 0, stream>>>(cb, cn);
    znorm_kernel<<<N4, 256, 0, stream>>>(act4, zn, T4);
    vq_gemm_kernel<<<dim3(VOCAB / 64, (N4 + 127) / 128), blk, 0, stream>>>(
        cb, act4, cn, Smat, T4, N4);
    init_counts_kernel<<<1, VOCAB, 0, stream>>>(counts);
    argmin_kernel<<<N4, 256, 0, stream>>>(Smat, zn, cb, counts, dmin,
                                          outZq, outEnc, outIdx, T4);
    finalize_kernel<<<1, 256, 0, stream>>>(dmin, counts, outLoss, outPerp, N4);
}
